// DecoderLayer_10033043604124
// MI455X (gfx1250) — matrix operand order repaired and checked by reading, not compiled, not run on hardware
//
#include <hip/hip_runtime.h>
#include <stdint.h>

// ---------------------------------------------------------------------------
// MI455X (gfx1250, wave32) fused decoder layer.
//   B=8, S=4096, D=1024, H=16, DH=64, F=4096
// Heavy path: fused KV-projection + flash-decode over 64-key chunks, bf16 WMMA.
// Light path: fp32 M=8 GEMMs + LayerNorm kernels (memory bound on weights).
// ---------------------------------------------------------------------------

typedef __attribute__((ext_vector_type(16))) __bf16 v16bf;
typedef __attribute__((ext_vector_type(8)))  float  v8f;

#define BQ   8
#define SEQ  4096
#define DIM  1024
#define NH   16
#define DH   64
#define FF   4096
#define CK   64                 // keys per chunk
#define NCH  (SEQ / CK)         // 64 chunks per batch row
#define STA  1032               // padded LDS row stride (bf16 elems): 2064B, 16B-aligned

union Frag16 { v16bf v; uint4 q[2]; };

__device__ __forceinline__ uint32_t pk_bf16x2(float lo, float hi) {
  uint32_t a = __float_as_uint(lo);
  uint32_t b = __float_as_uint(hi);
  a = (a + 0x7FFFu + ((a >> 16) & 1u)) >> 16;
  b = (b + 0x7FFFu + ((b >> 16) & 1u)) >> 16;
  return (a & 0xFFFFu) | (b << 16);
}
__device__ __forceinline__ uint16_t f2bf(float f) {
  uint32_t u = __float_as_uint(f);
  return (uint16_t)((u + 0x7FFFu + ((u >> 16) & 1u)) >> 16);
}
// cross-lane fetch from lane (lane ^ mask) within wave32
__device__ __forceinline__ float xget(float v, int mask) {
  int src = (((int)(threadIdx.x & 31)) ^ mask) << 2;
  return __int_as_float(__builtin_amdgcn_ds_bpermute(src, __float_as_int(v)));
}

// ---------------------------------------------------------------------------
// Per-wave projection: C[64x64] (one head's N-slice) = A[64x1024] * WT^T
// A: bf16 in LDS, row-major stride STA. WT: bf16 global, [N=1024][K=1024].
// ---------------------------------------------------------------------------
__device__ __forceinline__ void proj_head(const uint16_t* __restrict__ A,
                                          const uint16_t* __restrict__ WT,
                                          int h, int lane, v8f c[4][4]) {
  const int half = (lane >> 4) & 1;
  const int l15  = lane & 15;
#pragma unroll 4
  for (int kt = 0; kt < 32; ++kt) {
    const int k0 = kt * 32;
    Frag16 a[4];
#pragma unroll
    for (int mt = 0; mt < 4; ++mt) {
      // A layout: lanes 0-15 -> K[k0..k0+7], lanes 16-31 -> K[k0+8..k0+15] in q[0];
      //           +16 in q[1].  Row = mt*16 + (lane&15).
      const uint16_t* base = A + (size_t)(mt * 16 + l15) * STA + k0 + half * 8;
      a[mt].q[0] = *(const uint4*)(base);
      a[mt].q[1] = *(const uint4*)(base + 16);
    }
#pragma unroll
    for (int nt = 0; nt < 4; ++nt) {
      // B layout: col n = lane&15; lanes 0-15 -> K[k0..k0+7], lanes 16-31 -> K[k0+8..k0+15] in q[0]; +16 in q[1].
      const uint16_t* wb =
          WT + (size_t)(h * DH + nt * 16 + l15) * DIM + k0 + half * 8;
      Frag16 bfr;
      bfr.q[0] = *(const uint4*)(wb);
      bfr.q[1] = *(const uint4*)(wb + 16);
#pragma unroll
      for (int mt = 0; mt < 4; ++mt) {
        c[mt][nt] = __builtin_amdgcn_wmma_f32_16x16x32_bf16(
            false, a[mt].v, false, bfr.v, (short)0, c[mt][nt], false, false);
      }
    }
  }
}

// ---------------------------------------------------------------------------
// Fused KV projection + flash-decode partials.
// grid = B*NCH blocks, 512 threads (16 waves; wave w == head w).
// partials: per (b,h,chunk): [m, l, acc[64]] (66 floats)
// ---------------------------------------------------------------------------
__global__ void __launch_bounds__(512)
fused_kv_flash(const float* __restrict__ key, const float* __restrict__ value,
               const float* __restrict__ Q2, const uint16_t* __restrict__ WkT,
               const uint16_t* __restrict__ WvT, float* __restrict__ part) {
  extern __shared__ char smem[];
  uint16_t* Ak = (uint16_t*)smem;          // [CK][STA] bf16 key chunk
  uint16_t* Av = Ak + CK * STA;            // [CK][STA] bf16 value chunk

  const int tid  = threadIdx.x;
  const int lane = tid & 31;
  const int h    = tid >> 5;               // head == wave id
  const int blk  = blockIdx.x;
  const int b    = blk >> 6;               // / NCH
  const int c    = blk & (NCH - 1);

  // ---- stage key/value chunk into LDS as bf16 (512 thr, 8 thr per row) ----
  {
    const int r  = tid >> 3;               // 0..63
    const int cb = (tid & 7) * 128;        // column base
    const size_t rowoff = ((size_t)(b * SEQ + c * CK + r)) * DIM + cb;
    const float4* kp = (const float4*)(key + rowoff);
    const float4* vp = (const float4*)(value + rowoff);
    uint2* ok = (uint2*)(Ak + (size_t)r * STA + cb);
    uint2* ov = (uint2*)(Av + (size_t)r * STA + cb);
#pragma unroll 4
    for (int i = 0; i < 32; ++i) {
      float4 kk = kp[i];
      float4 vv = vp[i];
      ok[i] = make_uint2(pk_bf16x2(kk.x, kk.y), pk_bf16x2(kk.z, kk.w));
      ov[i] = make_uint2(pk_bf16x2(vv.x, vv.y), pk_bf16x2(vv.z, vv.w));
    }
  }
  __syncthreads();

  const int l15 = lane & 15;

  // ---- K projection (bias bk drops out of softmax) ----
  v8f cK[4][4];
#pragma unroll
  for (int mt = 0; mt < 4; ++mt)
#pragma unroll
    for (int nt = 0; nt < 4; ++nt) cK[mt][nt] = (v8f){0,0,0,0,0,0,0,0};
  proj_head(Ak, WkT, h, lane, cK);

  // q2 per lane (scaled by 1/sqrt(D) = 1/32)
  float q2v[4];
#pragma unroll
  for (int nt = 0; nt < 4; ++nt)
    q2v[nt] = Q2[b * DIM + h * DH + nt * 16 + l15] * 0.03125f;

  // scores: s[mt][j] over this lane's columns, then butterfly-sum over 16 lanes
  float w[4][8];
#pragma unroll
  for (int mt = 0; mt < 4; ++mt)
#pragma unroll
    for (int j = 0; j < 8; ++j)
      w[mt][j] = cK[mt][0][j] * q2v[0] + cK[mt][1][j] * q2v[1] +
                 cK[mt][2][j] * q2v[2] + cK[mt][3][j] * q2v[3];
#pragma unroll
  for (int mask = 1; mask < 16; mask <<= 1)
#pragma unroll
    for (int mt = 0; mt < 4; ++mt)
#pragma unroll
      for (int j = 0; j < 8; ++j) w[mt][j] += xget(w[mt][j], mask);

  // online-softmax partial for this chunk (rows split lanes<16 / lanes>=16)
  float ml = -3.0e38f;
#pragma unroll
  for (int mt = 0; mt < 4; ++mt)
#pragma unroll
    for (int j = 0; j < 8; ++j) ml = fmaxf(ml, w[mt][j]);
  const float mc = fmaxf(ml, xget(ml, 16));
  float ll = 0.f;
#pragma unroll
  for (int mt = 0; mt < 4; ++mt)
#pragma unroll
    for (int j = 0; j < 8; ++j) { w[mt][j] = __expf(w[mt][j] - mc); ll += w[mt][j]; }
  const float lc = ll + xget(ll, 16);

  // ---- V projection + weighted accumulation (bv added in reduce) ----
  v8f cV[4][4];
#pragma unroll
  for (int mt = 0; mt < 4; ++mt)
#pragma unroll
    for (int nt = 0; nt < 4; ++nt) cV[mt][nt] = (v8f){0,0,0,0,0,0,0,0};
  proj_head(Av, WvT, h, lane, cV);

  float o[4] = {0.f, 0.f, 0.f, 0.f};
#pragma unroll
  for (int mt = 0; mt < 4; ++mt)
#pragma unroll
    for (int nt = 0; nt < 4; ++nt)
#pragma unroll
      for (int j = 0; j < 8; ++j) o[nt] += w[mt][j] * cV[mt][nt][j];
#pragma unroll
  for (int nt = 0; nt < 4; ++nt) o[nt] += xget(o[nt], 16);   // join row halves

  float* p = part + ((size_t)((b * NH + h) * NCH + c)) * 66;
  if (lane == 0) { p[0] = mc; p[1] = lc; }
  if (lane < 16) {
#pragma unroll
    for (int nt = 0; nt < 4; ++nt) p[2 + nt * 16 + lane] = o[nt];
  }
}

// ---------------------------------------------------------------------------
// Combine per-chunk softmax partials -> attn_out[b, h*64+d]; adds bv.
// grid = B*NH blocks of 64 threads.
// ---------------------------------------------------------------------------
__global__ void attn_reduce(const float* __restrict__ part,
                            const float* __restrict__ bv,
                            float* __restrict__ attn_out) {
  const int bh = blockIdx.x;
  const int b = bh >> 4, h = bh & 15;
  const int t = threadIdx.x;                       // 0..63 = head dim
  const float* p0 = part + (size_t)bh * NCH * 66;
  float M = -3.0e38f;
  for (int c = 0; c < NCH; ++c) M = fmaxf(M, p0[c * 66]);
  float L = 0.f, acc = 0.f;
  for (int c = 0; c < NCH; ++c) {
    const float s = __expf(p0[c * 66] - M);
    L   += s * p0[c * 66 + 1];
    acc += s * p0[c * 66 + 2 + t];
  }
  attn_out[b * DIM + h * DH + t] = acc / L + bv[h * DH + t];
}

// ---------------------------------------------------------------------------
// Small fp32 GEMM: out[8][N] = act(in[8][K] @ W[K][N] + bias). Memory bound
// on W (read once, coalesced); `in` staged in LDS and reused for 8 rows.
// ---------------------------------------------------------------------------
__global__ void gemm8(const float* __restrict__ in, const float* __restrict__ W,
                      const float* __restrict__ bias, float* __restrict__ out,
                      int K, int N, int relu) {
  extern __shared__ float s_in[];                  // 8*K floats
  for (int i = threadIdx.x; i < 8 * K; i += blockDim.x) s_in[i] = in[i];
  __syncthreads();
  const int n = blockIdx.x * blockDim.x + threadIdx.x;
  if (n >= N) return;
  float acc[8] = {0, 0, 0, 0, 0, 0, 0, 0};
  for (int k = 0; k < K; ++k) {
    const float wv = W[(size_t)k * N + n];
#pragma unroll
    for (int r = 0; r < 8; ++r) acc[r] += s_in[r * K + k] * wv;
  }
  const float bb = bias[n];
#pragma unroll
  for (int r = 0; r < 8; ++r) {
    float v = acc[r] + bb;
    if (relu) v = fmaxf(v, 0.f);
    out[(size_t)r * N + n] = v;
  }
}

// ---------------------------------------------------------------------------
// out[r] = LayerNorm(a[r] + b[r]) * g + beta   (rows of 1024; grid=8, 256 thr)
// ---------------------------------------------------------------------------
__global__ void ln_add(const float* __restrict__ a, const float* __restrict__ bsrc,
                       const float* __restrict__ g, const float* __restrict__ beta,
                       float* __restrict__ out) {
  __shared__ float red[256];
  const int r = blockIdx.x, t = threadIdx.x;
  const float4 av = ((const float4*)(a + (size_t)r * DIM))[t];
  const float4 bv = ((const float4*)(bsrc + (size_t)r * DIM))[t];
  float x[4] = {av.x + bv.x, av.y + bv.y, av.z + bv.z, av.w + bv.w};

  float s = x[0] + x[1] + x[2] + x[3];
  red[t] = s; __syncthreads();
  for (int off = 128; off > 0; off >>= 1) {
    if (t < off) red[t] += red[t + off];
    __syncthreads();
  }
  const float mean = red[0] * (1.f / DIM);
  __syncthreads();

  float vs = 0.f;
#pragma unroll
  for (int i = 0; i < 4; ++i) { float d = x[i] - mean; vs += d * d; }
  red[t] = vs; __syncthreads();
  for (int off = 128; off > 0; off >>= 1) {
    if (t < off) red[t] += red[t + off];
    __syncthreads();
  }
  const float inv = rsqrtf(red[0] * (1.f / DIM) + 1e-6f);

  const float4 gv = ((const float4*)g)[t];
  const float4 pv = ((const float4*)beta)[t];
  float4 ov;
  ov.x = (x[0] - mean) * inv * gv.x + pv.x;
  ov.y = (x[1] - mean) * inv * gv.y + pv.y;
  ov.z = (x[2] - mean) * inv * gv.z + pv.z;
  ov.w = (x[3] - mean) * inv * gv.w + pv.w;
  ((float4*)(out + (size_t)r * DIM))[t] = ov;
}

// ---------------------------------------------------------------------------
// WT[n][k] = bf16(W[k][n]) : transpose-convert so B-fragments are contiguous.
// grid (32,32), block (32,8)
// ---------------------------------------------------------------------------
__global__ void wt_conv(const float* __restrict__ W, uint16_t* __restrict__ WT) {
  __shared__ float tile[32][33];
  const int nb = blockIdx.x * 32, kb = blockIdx.y * 32;
  const int tx = threadIdx.x, ty = threadIdx.y;
#pragma unroll
  for (int i = 0; i < 32; i += 8)
    tile[ty + i][tx] = W[(size_t)(kb + ty + i) * DIM + nb + tx];
  __syncthreads();
#pragma unroll
  for (int i = 0; i < 32; i += 8)
    WT[(size_t)(nb + ty + i) * DIM + kb + tx] = f2bf(tile[tx][ty + i]);
}

// ---------------------------------------------------------------------------
extern "C" void kernel_launch(void* const* d_in, const int* in_sizes, int n_in,
                              void* d_out, int out_size, void* d_ws, size_t ws_size,
                              hipStream_t stream) {
  const float* key   = (const float*)d_in[0];
  const float* value = (const float*)d_in[1];
  const float* x0    = (const float*)d_in[2];   // decode_input [8,1,1024]
  const float* Wq = (const float*)d_in[3];  const float* bq = (const float*)d_in[4];
  const float* Wk = (const float*)d_in[5];  /* bk drops out of softmax */
  const float* Wv = (const float*)d_in[7];  const float* bv = (const float*)d_in[8];
  const float* Wo = (const float*)d_in[9];  const float* bo = (const float*)d_in[10];
  const float* W1 = (const float*)d_in[11]; const float* b1 = (const float*)d_in[12];
  const float* W2 = (const float*)d_in[13]; const float* b2 = (const float*)d_in[14];
  const float* ln2_g = (const float*)d_in[15]; const float* ln2_b = (const float*)d_in[16];
  const float* lnf_g = (const float*)d_in[17]; const float* lnf_b = (const float*)d_in[18];

  char* ws = (char*)d_ws;
  uint16_t* WkT = (uint16_t*)ws;                              // 2 MB
  uint16_t* WvT = (uint16_t*)(ws + (2u << 20));               // 2 MB
  float* V1   = (float*)(ws + (4u << 20));                    // 8x1024 each
  float* mha1 = V1 + BQ * DIM;
  float* Q2   = mha1 + BQ * DIM;
  float* attn = Q2 + BQ * DIM;
  float* mha2 = attn + BQ * DIM;
  float* xln  = mha2 + BQ * DIM;
  float* h1   = xln + BQ * DIM;                               // 8x4096
  float* yff  = h1 + BQ * FF;
  float* part = yff + BQ * DIM;                               // 8*16*64*66 floats

  // weight prep (bf16, transposed) — independent of the q-chain
  wt_conv<<<dim3(32, 32), dim3(32, 8), 0, stream>>>(Wk, WkT);
  wt_conv<<<dim3(32, 32), dim3(32, 8), 0, stream>>>(Wv, WvT);

  // Q2 chain: mha1 has S=1 => softmax==1 => mha1 = (x0@Wv+bv)@Wo+bo
  const size_t sh8k = (size_t)8 * DIM * sizeof(float);
  gemm8<<<(DIM + 255) / 256, 256, sh8k, stream>>>(x0,   Wv, bv, V1,   DIM, DIM, 0);
  gemm8<<<(DIM + 255) / 256, 256, sh8k, stream>>>(V1,   Wo, bo, mha1, DIM, DIM, 0);
  gemm8<<<(DIM + 255) / 256, 256, sh8k, stream>>>(mha1, Wq, bq, Q2,   DIM, DIM, 0);

  // fused KV projection + flash-decode partials (the 137-GFLOP WMMA path)
  const size_t shBig = (size_t)2 * CK * STA * sizeof(uint16_t);  // 264192 B LDS
  fused_kv_flash<<<BQ * NCH, 512, shBig, stream>>>(key, value, Q2, WkT, WvT, part);
  attn_reduce<<<BQ * NH, 64, 0, stream>>>(part, bv, attn);

  // epilogue: Wo, LN, FFN, LN
  gemm8<<<(DIM + 255) / 256, 256, sh8k, stream>>>(attn, Wo, bo, mha2, DIM, DIM, 0);
  ln_add<<<BQ, 256, 0, stream>>>(mha2, x0, ln2_g, ln2_b, xln);
  gemm8<<<(FF + 255) / 256, 256, sh8k, stream>>>(xln, W1, b1, h1, DIM, FF, 1);
  gemm8<<<(DIM + 255) / 256, 256, (size_t)8 * FF * sizeof(float), stream>>>(
      h1, W2, b2, yff, FF, DIM, 0);
  ln_add<<<BQ, 256, 0, stream>>>(yff, xln, lnf_g, lnf_b, (float*)d_out);
}
